// Attention_9577777070737
// MI455X (gfx1250) — matrix operand order repaired and checked by reading, not compiled, not run on hardware
//
#include <hip/hip_runtime.h>
#include <hip/hip_bf16.h>

typedef __bf16 bf16_t;
typedef bf16_t v16bf __attribute__((ext_vector_type(16)));
typedef float  v8f   __attribute__((ext_vector_type(8)));
typedef int    v4i_t __attribute__((vector_size(16)));

#define WIN_N      197
#define NPAD       224        // padded sequence length (14 * 16)
#define HEADS      16
#define HD         64
#define B_BATCH    64
#define C_DIM      1024
#define M_TOTAL    (B_BATCH * WIN_N)   // 12608
#define NN         (WIN_N * WIN_N)     // 38809
#define S_STRIDE   228        // f32 score row stride (mult of 4)
#define QK_STRIDE  72         // bf16 Q/K LDS row stride (mult of 8)

#if defined(__has_builtin)
#if __has_builtin(__builtin_amdgcn_global_load_async_to_lds_b128)
#define HAS_ASYNC_LDS 1
#endif
#endif
#ifndef HAS_ASYNC_LDS
#define HAS_ASYNC_LDS 0
#endif

#if HAS_ASYNC_LDS
typedef __attribute__((address_space(1))) v4i_t gv4i;
typedef __attribute__((address_space(3))) v4i_t lv4i;
#endif

// 16-byte global -> LDS copy; async (ASYNCcnt-tracked, no VGPR round trip)
// when the gfx1250 builtin is available, plain copy otherwise.
__device__ inline void copy16_g2l(const void* gptr, void* lptr) {
#if HAS_ASYNC_LDS
    __builtin_amdgcn_global_load_async_to_lds_b128(
        (gv4i*)(v4i_t*)gptr, (lv4i*)(v4i_t*)lptr, 0, 0);
#else
    *reinterpret_cast<uint4*>(lptr) = *reinterpret_cast<const uint4*>(gptr);
#endif
}

__device__ inline void wait_async_copies() {
#if HAS_ASYNC_LDS
#if defined(__has_builtin) && __has_builtin(__builtin_amdgcn_s_wait_asynccnt)
    __builtin_amdgcn_s_wait_asynccnt(0);
#else
    asm volatile("s_wait_asynccnt 0" ::: "memory");
#endif
#endif
}

__device__ inline v8f zero_v8f() {
    v8f z = {0.f,0.f,0.f,0.f,0.f,0.f,0.f,0.f};
    return z;
}

__device__ inline v8f wmma_bf16(v16bf a, v16bf b, v8f c) {
    return __builtin_amdgcn_wmma_f32_16x16x32_bf16(
        /*neg_a=*/false, a, /*neg_b=*/false, b,
        /*c_mod=*/(short)0, c, /*reuse_a=*/false, /*reuse_b=*/false);
}

// A fragment (16 x 32 bf16, row-major in LDS). base -> tile origin (row0,k0).
// lane<16: K {0..7, 16..23}; lane>=16: K {8..15, 24..31} (ISA 7.12.2).
__device__ inline v16bf load_a_bf16(const bf16_t* base, int stride, int lane) {
    const int half = lane >> 4, r = lane & 15;
    const bf16_t* p = base + r * stride + half * 8;
    union { v16bf v; struct { uint4 x, y; } q; } u;
    u.q.x = *reinterpret_cast<const uint4*>(p);
    u.q.y = *reinterpret_cast<const uint4*>(p + 16);
    return u.v;
}

// B fragment (32 k x 16 n), stored transposed (N-major: Bt[n][k]).
// lane = column n (&15); lane<16: K {0..7, 16..23}; lane>=16: K {8..15, 24..31}.
__device__ inline v16bf load_b_bf16(const bf16_t* base, int stride, int lane) {
    const int half = lane >> 4, n = lane & 15;
    const bf16_t* p = base + n * stride + half * 8;
    union { v16bf v; struct { uint4 x, y; } q; } u;
    u.q.x = *reinterpret_cast<const uint4*>(p);
    u.q.y = *reinterpret_cast<const uint4*>(p + 16);
    return u.v;
}

// A fragment built from f32 LDS (softmax probabilities), converted to bf16.
__device__ inline v16bf load_a_f32(const float* base, int stride, int lane) {
    const int half = lane >> 4, r = lane & 15;
    const float* p0 = base + r * stride + half * 8;
    const float* p1 = base + r * stride + 16 + half * 8;
    v16bf a;
#pragma unroll
    for (int i = 0; i < 8; ++i) a[i] = (bf16_t)p0[i];
#pragma unroll
    for (int i = 0; i < 8; ++i) a[8 + i] = (bf16_t)p1[i];
    return a;
}

// ---------------------------------------------------------------------------
// Kernel 1: gather relative-position bias -> biasb[h][q*197+k] (f32)
// ---------------------------------------------------------------------------
__global__ void bias_gather_kernel(const float* __restrict__ rpb,
                                   const int* __restrict__ relidx,
                                   float* __restrict__ biasb) {
    int i = blockIdx.x * blockDim.x + threadIdx.x;
    if (i >= NN) return;
    int idx = relidx[i];
#pragma unroll
    for (int h = 0; h < HEADS; ++h)
        biasb[(size_t)h * NN + i] = rpb[idx * HEADS + h];
}

// ---------------------------------------------------------------------------
// Kernel 2: QKV GEMM.  qkv[m][n] = x[m][:] . qkv_w[n][:]  (+ bias, routed)
//   Q -> qbuf [BH][NPAD][HD] bf16 (scaled by 1/8), K -> kbuf same layout,
//   V -> vtbuf [BH][HD][NPAD] bf16 (transposed for B-fragment loads)
// ---------------------------------------------------------------------------
__global__ __launch_bounds__(256) void qkv_gemm_kernel(
    const float* __restrict__ x, const float* __restrict__ qkv_w,
    const float* __restrict__ q_bias, const float* __restrict__ v_bias,
    bf16_t* __restrict__ qbuf, bf16_t* __restrict__ kbuf,
    bf16_t* __restrict__ vtbuf) {
    __shared__ bf16_t As[128 * 40];
    __shared__ bf16_t Bs[128 * 40];
    const int tid = threadIdx.x, lane = tid & 31, wave = tid >> 5;
    const int m0 = blockIdx.x * 128, n0 = blockIdx.y * 128;

    v8f acc[8];
#pragma unroll
    for (int i = 0; i < 8; ++i) acc[i] = zero_v8f();

    for (int k0 = 0; k0 < C_DIM; k0 += 32) {
        // stage A (x, f32 -> bf16): 128x32, 4 float4 chunks per thread
#pragma unroll
        for (int c = 0; c < 4; ++c) {
            int idx = tid + c * 256;
            int row = idx >> 3, col = (idx & 7) * 4;
            int gm = m0 + row;
            float4 f = make_float4(0.f, 0.f, 0.f, 0.f);
            if (gm < M_TOTAL)
                f = *reinterpret_cast<const float4*>(x + (size_t)gm * C_DIM + k0 + col);
            bf16_t* d = As + row * 40 + col;
            d[0] = (bf16_t)f.x; d[1] = (bf16_t)f.y;
            d[2] = (bf16_t)f.z; d[3] = (bf16_t)f.w;
        }
        // stage B (qkv_w rows = output columns, natural N-major x K layout)
#pragma unroll
        for (int c = 0; c < 4; ++c) {
            int idx = tid + c * 256;
            int row = idx >> 3, col = (idx & 7) * 4;
            float4 f = *reinterpret_cast<const float4*>(
                qkv_w + (size_t)(n0 + row) * C_DIM + k0 + col);
            bf16_t* d = Bs + row * 40 + col;
            d[0] = (bf16_t)f.x; d[1] = (bf16_t)f.y;
            d[2] = (bf16_t)f.z; d[3] = (bf16_t)f.w;
        }
        __syncthreads();
        v16bf a = load_a_bf16(As + wave * 16 * 40, 40, lane);
#pragma unroll
        for (int nt = 0; nt < 8; ++nt) {
            v16bf b = load_b_bf16(Bs + nt * 16 * 40, 40, lane);
            acc[nt] = wmma_bf16(a, b, acc[nt]);
        }
        __syncthreads();
    }

    const int half = lane >> 4, ln = lane & 15;
#pragma unroll
    for (int nt = 0; nt < 8; ++nt) {
#pragma unroll
        for (int r = 0; r < 8; ++r) {
            int gm = m0 + wave * 16 + r + 8 * half;
            int gn = n0 + nt * 16 + ln;
            if (gm >= M_TOTAL) continue;
            int batch = gm / WIN_N, n = gm - batch * WIN_N;
            int sel = gn >> 10, cc = gn & 1023, h = cc >> 6, d = cc & 63;
            size_t bh = (size_t)batch * HEADS + h;
            float v = acc[nt][r];
            if (sel == 0) {
                v = (v + q_bias[cc]) * 0.125f;  // SCALE = 64^-0.5
                qbuf[(bh * NPAD + n) * HD + d] = (bf16_t)v;
            } else if (sel == 1) {
                kbuf[(bh * NPAD + n) * HD + d] = (bf16_t)v;
            } else {
                v += v_bias[cc];
                vtbuf[(bh * HD + d) * NPAD + n] = (bf16_t)v;
            }
        }
    }
}

// ---------------------------------------------------------------------------
// Kernel 3: attention per (b,h). Full 197x197 score matrix in LDS (f32),
// QK^T and P@V via bf16 WMMA, row softmax in f32. Staging uses async
// global->LDS copies (ASYNCcnt) when available.
// ---------------------------------------------------------------------------
__global__ __launch_bounds__(256) void attention_kernel(
    const bf16_t* __restrict__ qbuf, const bf16_t* __restrict__ kbuf,
    const bf16_t* __restrict__ vtbuf, const float* __restrict__ biasb,
    bf16_t* __restrict__ ctx) {
    extern __shared__ char smem[];
    bf16_t* Qs  = (bf16_t*)smem;                 // [NPAD][QK_STRIDE]
    bf16_t* Ks  = Qs + NPAD * QK_STRIDE;         // [NPAD][QK_STRIDE]
    bf16_t* Vts = Ks + NPAD * QK_STRIDE;         // [HD][NPAD]
    float*  S   = (float*)(Vts + HD * NPAD);     // [NPAD][S_STRIDE]

    const int tid = threadIdx.x, lane = tid & 31, wave = tid >> 5;
    const int bh = blockIdx.x;
    const int batch = bh >> 4, h = bh & 15;
    const bf16_t* qg = qbuf + (size_t)bh * NPAD * HD;
    const bf16_t* kg = kbuf + (size_t)bh * NPAD * HD;
    const bf16_t* vg = vtbuf + (size_t)bh * HD * NPAD;
    const float*  bg = biasb + (size_t)h * NN;

    // stage Q,K (row stride 64 -> 72) and Vt (flat copy), 16B async chunks
    for (int c = tid; c < NPAD * (HD / 8); c += 256) {
        int row = c >> 3, col = (c & 7) * 8;
        copy16_g2l(qg + row * HD + col, Qs + row * QK_STRIDE + col);
        copy16_g2l(kg + row * HD + col, Ks + row * QK_STRIDE + col);
    }
    for (int c = tid; c < HD * NPAD / 8; c += 256)
        copy16_g2l(vg + c * 8, Vts + c * 8);
    wait_async_copies();
    __syncthreads();

    const int half = lane >> 4, ln = lane & 15;

    // S = Q @ K^T (+bias, masked). 13 q-tiles x 14 k-tiles, K-dim 64 = 2 WMMAs
    for (int job = wave; job < 13 * 14; job += 8) {
        int qt = job / 14, kt = job - qt * 14;
        v16bf a0 = load_a_bf16(Qs + qt * 16 * QK_STRIDE, QK_STRIDE, lane);
        v16bf a1 = load_a_bf16(Qs + qt * 16 * QK_STRIDE + 32, QK_STRIDE, lane);
        v16bf b0 = load_b_bf16(Ks + kt * 16 * QK_STRIDE, QK_STRIDE, lane);
        v16bf b1 = load_b_bf16(Ks + kt * 16 * QK_STRIDE + 32, QK_STRIDE, lane);
        v8f c = zero_v8f();
        c = wmma_bf16(a0, b0, c);
        c = wmma_bf16(a1, b1, c);
#pragma unroll
        for (int r = 0; r < 8; ++r) {
            int q = qt * 16 + r + 8 * half;
            int k = kt * 16 + ln;
            float val;
            if (q < WIN_N && k < WIN_N)      val = c[r] + bg[q * WIN_N + k];
            else if (q < WIN_N)              val = -1e30f;  // padded key
            else                             val = 0.f;     // padded query row
            S[q * S_STRIDE + k] = val;
        }
    }
    __syncthreads();

    // row softmax (f32), one thread per row
    if (tid < WIN_N) {
        float* row = S + tid * S_STRIDE;
        float m = -1e30f;
        for (int j = 0; j < WIN_N; ++j) m = fmaxf(m, row[j]);
        float s = 0.f;
        for (int j = 0; j < WIN_N; ++j) {
            float e = __expf(row[j] - m);
            row[j] = e;
            s += e;
        }
        float inv = 1.f / s;
        for (int j = 0; j < WIN_N; ++j) row[j] *= inv;
        for (int j = WIN_N; j < NPAD; ++j) row[j] = 0.f;  // zero padded keys
    }
    __syncthreads();

    // O = P @ V : 13 q-tiles x 4 d-tiles, K-dim 224 = 7 WMMAs
    for (int job = wave; job < 13 * 4; job += 8) {
        int qt = job >> 2, dt = job & 3;
        v8f c = zero_v8f();
#pragma unroll
        for (int ks = 0; ks < 7; ++ks) {
            v16bf a = load_a_f32(S + qt * 16 * S_STRIDE + ks * 32, S_STRIDE, lane);
            v16bf b = load_b_bf16(Vts + dt * 16 * NPAD + ks * 32, NPAD, lane);
            c = wmma_bf16(a, b, c);
        }
#pragma unroll
        for (int r = 0; r < 8; ++r) {
            int q = qt * 16 + r + 8 * half;
            int d = dt * 16 + ln;
            if (q < WIN_N)
                ctx[((size_t)batch * WIN_N + q) * C_DIM + h * HD + d] = (bf16_t)c[r];
        }
    }
}

// ---------------------------------------------------------------------------
// Kernel 4: output projection.  out[m][n] = ctx[m][:] . proj_w[n][:] + proj_b
// A-stage (bf16 ctx) uses async global->LDS; out-of-range rows are clamped to
// a valid source row (their D rows are never stored).
// ---------------------------------------------------------------------------
__global__ __launch_bounds__(256) void proj_gemm_kernel(
    const bf16_t* __restrict__ ctx, const float* __restrict__ proj_w,
    const float* __restrict__ proj_b, float* __restrict__ out) {
    __shared__ bf16_t As[128 * 40];
    __shared__ bf16_t Bs[128 * 40];
    const int tid = threadIdx.x, lane = tid & 31, wave = tid >> 5;
    const int m0 = blockIdx.x * 128, n0 = blockIdx.y * 128;

    v8f acc[8];
#pragma unroll
    for (int i = 0; i < 8; ++i) acc[i] = zero_v8f();

    for (int k0 = 0; k0 < C_DIM; k0 += 32) {
        // stage A (ctx already bf16): 512 async 16B chunks, 2 per thread
#pragma unroll
        for (int c = 0; c < 2; ++c) {
            int idx = tid + c * 256;
            int row = idx >> 2, col = (idx & 3) * 8;
            int gm = m0 + row;
            if (gm >= M_TOTAL) gm = M_TOTAL - 1;  // clamp; dead rows not stored
            copy16_g2l(ctx + (size_t)gm * C_DIM + k0 + col, As + row * 40 + col);
        }
        // stage B (proj_w f32 -> bf16)
#pragma unroll
        for (int c = 0; c < 4; ++c) {
            int idx = tid + c * 256;
            int row = idx >> 3, col = (idx & 7) * 4;
            float4 f = *reinterpret_cast<const float4*>(
                proj_w + (size_t)(n0 + row) * C_DIM + k0 + col);
            bf16_t* d = Bs + row * 40 + col;
            d[0] = (bf16_t)f.x; d[1] = (bf16_t)f.y;
            d[2] = (bf16_t)f.z; d[3] = (bf16_t)f.w;
        }
        wait_async_copies();
        __syncthreads();
        v16bf a = load_a_bf16(As + wave * 16 * 40, 40, lane);
#pragma unroll
        for (int nt = 0; nt < 8; ++nt) {
            v16bf b = load_b_bf16(Bs + nt * 16 * 40, 40, lane);
            acc[nt] = wmma_bf16(a, b, acc[nt]);
        }
        __syncthreads();
    }

    const int half = lane >> 4, ln = lane & 15;
#pragma unroll
    for (int nt = 0; nt < 8; ++nt) {
#pragma unroll
        for (int r = 0; r < 8; ++r) {
            int gm = m0 + wave * 16 + r + 8 * half;
            int gn = n0 + nt * 16 + ln;
            if (gm < M_TOTAL)
                out[(size_t)gm * C_DIM + gn] = acc[nt][r] + proj_b[gn];
        }
    }
}

// ---------------------------------------------------------------------------
extern "C" void kernel_launch(void* const* d_in, const int* in_sizes, int n_in,
                              void* d_out, int out_size, void* d_ws, size_t ws_size,
                              hipStream_t stream) {
    (void)in_sizes; (void)n_in; (void)out_size; (void)ws_size;
    const float* x       = (const float*)d_in[0];
    const float* qkv_w   = (const float*)d_in[1];
    const float* q_bias  = (const float*)d_in[2];
    const float* v_bias  = (const float*)d_in[3];
    const float* rpb     = (const float*)d_in[4];
    const float* proj_w  = (const float*)d_in[5];
    const float* proj_b  = (const float*)d_in[6];
    const int*   relidx  = (const int*)d_in[7];
    float* out = (float*)d_out;

    // workspace carve (256B aligned)
    const size_t qkv_bytes = (size_t)B_BATCH * HEADS * NPAD * HD * sizeof(bf16_t); // 29.36 MB
    const size_t ctx_bytes  = (size_t)M_TOTAL * C_DIM * sizeof(bf16_t);
    const size_t bias_bytes = (size_t)HEADS * NN * sizeof(float);
    char* p = (char*)d_ws;
    bf16_t* qbuf  = (bf16_t*)p; p += qkv_bytes;
    bf16_t* kbuf  = (bf16_t*)p; p += qkv_bytes;
    bf16_t* vtbuf = (bf16_t*)p; p += qkv_bytes;
    bf16_t* ctx   = (bf16_t*)p; p += (ctx_bytes + 255) & ~(size_t)255;
    float*  biasb = (float*)p;  p += (bias_bytes + 255) & ~(size_t)255;

    // zero padded rows/cols of Q/K/Vt (pad keys must be finite zeros)
    (void)hipMemsetAsync(qbuf, 0, 3 * qkv_bytes, stream);

    bias_gather_kernel<<<(NN + 255) / 256, 256, 0, stream>>>(rpb, relidx, biasb);

    dim3 g1((M_TOTAL + 127) / 128, 3 * C_DIM / 128);  // 99 x 24
    qkv_gemm_kernel<<<g1, 256, 0, stream>>>(x, qkv_w, q_bias, v_bias,
                                            qbuf, kbuf, vtbuf);

    const size_t smem_att = (size_t)(2 * NPAD * QK_STRIDE + HD * NPAD) * sizeof(bf16_t)
                          + (size_t)NPAD * S_STRIDE * sizeof(float);   // 297,472 B
    attention_kernel<<<B_BATCH * HEADS, 256, smem_att, stream>>>(
        qbuf, kbuf, vtbuf, biasb, ctx);

    dim3 g2((M_TOTAL + 127) / 128, C_DIM / 128);      // 99 x 8
    proj_gemm_kernel<<<g2, 256, 0, stream>>>(ctx, proj_w, proj_b, out);
}
